// Block_65146063946124
// MI455X (gfx1250) — matrix operand order repaired and checked by reading, not compiled, not run on hardware
//
#include <hip/hip_runtime.h>
#include <hip/hip_bf16.h>
#include <math.h>

typedef unsigned short u16;
typedef int v4i __attribute__((__vector_size__(16)));
typedef __attribute__((ext_vector_type(16))) __bf16 v16bf;
typedef __attribute__((ext_vector_type(8)))  float  v8f;

#define BSZ   2
#define TLEN  2048
#define CDIM  1024
#define C4    4096
#define NHEAD 16
#define HD    64
#define BT    (BSZ * TLEN)   // 4096 tokens

#ifndef __has_builtin
#define __has_builtin(x) 0
#endif

#if __has_builtin(__builtin_amdgcn_global_load_async_to_lds_b128)
#define HAS_ASYNC_LDS 1
#else
#define HAS_ASYNC_LDS 0
#endif

// ---------------------------------------------------------------- helpers ---

__device__ __forceinline__ u16 f2bf(float f) {
  unsigned int u = __float_as_uint(f);
  unsigned int r = (u >> 16) & 1u;            // round-to-nearest-even
  u += 0x7fffu + r;
  return (u16)(u >> 16);
}

// Copy 16 bytes global -> LDS. Uses the CDNA5 async-to-LDS path (ASYNCcnt
// tracked, no VGPR round trip) when the toolchain exposes the builtin.
// Builtin parameter types are generic `v4i*` (from the compiler diagnostic).
__device__ __forceinline__ void cp16(u16* lp, const u16* gp) {
#if HAS_ASYNC_LDS
  __builtin_amdgcn_global_load_async_to_lds_b128((v4i*)gp, (v4i*)lp, 0, 0);
#else
  *(uint4*)lp = *(const uint4*)gp;
#endif
}

__device__ __forceinline__ void cp_wait() {
#if HAS_ASYNC_LDS
#if __has_builtin(__builtin_amdgcn_s_wait_asynccnt)
  __builtin_amdgcn_s_wait_asynccnt(0);
#else
  asm volatile("s_wait_asynccnt 0x0" ::: "memory");
#endif
#endif
}

// A fragment (16x32 bf16, ISA 7.12.2) from a ROW-MAJOR LDS tile.
// Lane l: row m=l&15, half h=l>>4. Elements 0..15 -> K = 16h+0..15:
// one contiguous 32B run -> 2x ds_load_b128.
__device__ __forceinline__ v16bf load_a_frag_rm(const __bf16* p, int ld, int lane) {
  const int m = lane & 15;
  const int h = lane >> 4;
  const __bf16* q = p + m * ld + 16 * h;
  v16bf a;
#pragma unroll
  for (int e = 0; e < 8; ++e) a[e] = q[e];
#pragma unroll
  for (int e = 0; e < 8; ++e) a[8 + e] = q[8 + e];
  return a;
}

// B fragment where the per-lane 16 elements are CONTIGUOUS in LDS
// (tile stored so that (n, k) -> n*32 + k). Lane l: col n=l&15, half h
// selects K 0..15 / 16..31: one contiguous 32B run -> 2x ds_load_b128.
__device__ __forceinline__ v16bf load_frag16(const __bf16* p) {
  v16bf b;
#pragma unroll
  for (int e = 0; e < 16; ++e) b[e] = p[e];
  return b;
}

__device__ __forceinline__ void zero8(v8f& v) {
#pragma unroll
  for (int e = 0; e < 8; ++e) v[e] = 0.0f;
}

// -------------------------------------------------------- fp32 -> bf16 -----

__global__ __launch_bounds__(256) void f2bf4_kernel(const float4* __restrict__ in,
                                                    ushort4* __restrict__ out,
                                                    int n4) {
  for (int i = blockIdx.x * 256 + threadIdx.x; i < n4; i += gridDim.x * 256) {
    const float4 f = in[i];
    ushort4 o;
    o.x = f2bf(f.x); o.y = f2bf(f.y); o.z = f2bf(f.z); o.w = f2bf(f.w);
    out[i] = o;
  }
}

// ------------------------------------------------------------- layernorm ---

__global__ __launch_bounds__(256) void ln_kernel(const float* __restrict__ x,
                                                 const float* __restrict__ g,
                                                 const float* __restrict__ b,
                                                 float* __restrict__ outF,
                                                 u16* __restrict__ outB) {
  const int row = blockIdx.x;
  const int tid = threadIdx.x;
  const float* xr = x + (size_t)row * CDIM;
  __shared__ float red[256];

  float s = 0.f;
  for (int c = tid; c < CDIM; c += 256) s += xr[c];
  red[tid] = s;
  __syncthreads();
  for (int o = 128; o > 0; o >>= 1) {
    if (tid < o) red[tid] += red[tid + o];
    __syncthreads();
  }
  const float mean = red[0] * (1.0f / CDIM);
  __syncthreads();

  float vs = 0.f;
  for (int c = tid; c < CDIM; c += 256) {
    const float d = xr[c] - mean;
    vs += d * d;
  }
  red[tid] = vs;
  __syncthreads();
  for (int o = 128; o > 0; o >>= 1) {
    if (tid < o) red[tid] += red[tid + o];
    __syncthreads();
  }
  const float inv = rsqrtf(red[0] * (1.0f / CDIM) + 1e-5f);

  for (int c = tid; c < CDIM; c += 256) {
    const float y = (xr[c] - mean) * inv * g[c] + b[c];
    const size_t idx = (size_t)row * CDIM + c;
    outF[idx] = y;
    outB[idx] = f2bf(y);
  }
}

// ------------------------------------------------ bf16 WMMA tiled GEMM -----
// C[M,N] = A[M,K] @ B[K,N] (+resid) (relu?) -> fp32 and/or bf16 outputs.
// Block tile 128x128, K step 32; 8 waves arranged 4(M) x 2(N); each wave
// computes 32x64 as 2x4 v_wmma_f32_16x16x32_bf16 per K step.
// A staged row-major (async path); B staged TRANSPOSED [n][k] so every
// B fragment is a contiguous 32-byte LDS read per lane.

template <bool WF, bool WB, bool RESID, bool RELU>
__global__ __launch_bounds__(256) void gemm_bf16_kernel(
    const u16* __restrict__ A, const u16* __restrict__ Bw,
    const float* __restrict__ resid,
    float* __restrict__ outF, u16* __restrict__ outB,
    int M, int N, int K) {
  __shared__ __align__(16) u16 sA[128 * 32];   // row-major [m][k], 8 KB
  __shared__ __align__(16) u16 sBT[128 * 32];  // transposed [n][k], 8 KB

  const int tid  = threadIdx.x;
  const int lane = tid & 31;
  const int wid  = tid >> 5;   // 0..7
  const int wm   = wid >> 1;   // 0..3  (M direction, 32 rows each)
  const int wn   = wid & 1;    // 0..1  (N direction, 64 cols each)
  const int bm0  = blockIdx.x * 128;
  const int bn0  = blockIdx.y * 128;
  const int mlane = lane & 15;
  const int hlane = lane >> 4;

  v8f acc[2][4];
#pragma unroll
  for (int i = 0; i < 2; ++i)
#pragma unroll
    for (int j = 0; j < 4; ++j) zero8(acc[i][j]);

  for (int k0 = 0; k0 < K; k0 += 32) {
    __syncthreads();
    // A tile: 128x32 = 512 x 16B chunks, 2 per thread (async-to-LDS path).
#pragma unroll
    for (int v = tid; v < 512; v += 256) {
      const int row = v >> 2, part = v & 3;
      cp16(&sA[row * 32 + part * 8],
           &A[(size_t)(bm0 + row) * K + k0 + part * 8]);
    }
    // B tile transposed: read 8 consecutive n at fixed k, scatter to [n][k].
#pragma unroll
    for (int v = tid; v < 512; v += 256) {
      const int krow = v >> 4, npart = v & 15;
      const uint4 d = *(const uint4*)&Bw[(size_t)(k0 + krow) * N + bn0 + npart * 8];
      const u16* e = (const u16*)&d;
#pragma unroll
      for (int t = 0; t < 8; ++t)
        sBT[(npart * 8 + t) * 32 + krow] = e[t];
    }
    cp_wait();
    __syncthreads();

    const __bf16* pA  = reinterpret_cast<const __bf16*>(sA);
    const __bf16* pBT = reinterpret_cast<const __bf16*>(sBT);

    v16bf af[2], bfr[4];
#pragma unroll
    for (int i = 0; i < 2; ++i)
      af[i] = load_a_frag_rm(pA + (wm * 32 + i * 16) * 32, 32, lane);
#pragma unroll
    for (int j = 0; j < 4; ++j)
      bfr[j] = load_frag16(pBT + (wn * 64 + j * 16 + mlane) * 32 + 16 * hlane);

#pragma unroll
    for (int i = 0; i < 2; ++i)
#pragma unroll
      for (int j = 0; j < 4; ++j)
        acc[i][j] = __builtin_amdgcn_wmma_f32_16x16x32_bf16(
            false, af[i], false, bfr[j], (short)0, acc[i][j], false, false);
  }

  // Epilogue — C/D layout: lanes 0-15: M=r, lanes 16-31: M=8+r; N=lane&15.
#pragma unroll
  for (int i = 0; i < 2; ++i)
#pragma unroll
    for (int j = 0; j < 4; ++j)
#pragma unroll
      for (int r = 0; r < 8; ++r) {
        const int row = bm0 + wm * 32 + i * 16 + r + 8 * hlane;
        const int col = bn0 + wn * 64 + j * 16 + mlane;
        const size_t idx = (size_t)row * N + col;
        float v = acc[i][j][r];
        if (RESID) v += resid[idx];
        if (RELU) v = fmaxf(v, 0.0f);
        if (WF) outF[idx] = v;
        if (WB) outB[idx] = f2bf(v);
      }
}

// ------------------------------------------------- flash attention (bf16) --
// One block = one (batch, head, 64-query tile). 4 waves, each owns 16 query
// rows. Streams 32-key blocks through LDS; scores via Q·K^T WMMA; online
// softmax in fp32 (quirk: scores *= HEAD_DIM); P·V via WMMA. K tile stays
// row-major (K^T fragments are contiguous); V tile is stored transposed
// [dim][key] so P·V B-fragments are contiguous too.

__global__ __launch_bounds__(128) void attn_kernel(const u16* __restrict__ qb,
                                                   const u16* __restrict__ kb,
                                                   const u16* __restrict__ vb,
                                                   u16* __restrict__ ob) {
  const int qt  = blockIdx.x;   // query tile (64 rows)
  const int hhd = blockIdx.y;   // head
  const int bbb = blockIdx.z;   // batch
  const int tid = threadIdx.x;
  const int lane = tid & 31;
  const int wid  = tid >> 5;    // 0..3
  const int q0   = qt * 64;

  __shared__ __align__(16) u16 Qs[64 * HD];      // 8 KB row-major [q][dim]
  __shared__ __align__(16) u16 Ks[32 * HD];      // 4 KB row-major [key][dim]
  __shared__ __align__(16) u16 VsT[HD * 32];     // 4 KB transposed [dim][key]
  __shared__ __align__(16) u16 Ps[4][16 * 32];   // 4 KB per-wave P tiles

  const size_t base = ((size_t)bbb * TLEN) * CDIM + (size_t)hhd * HD;

  // Q tile: 64x64 = 512 x 16B chunks, 4 per thread (async path).
#pragma unroll
  for (int v = tid; v < 512; v += 128) {
    const int row = v >> 3, part = v & 7;
    cp16(&Qs[row * HD + part * 8],
         &qb[base + (size_t)(q0 + row) * CDIM + part * 8]);
  }
  cp_wait();
  __syncthreads();

  const __bf16* Qp = reinterpret_cast<const __bf16*>(Qs);
  const v16bf aQ0 = load_a_frag_rm(Qp + (wid * 16) * HD + 0, HD, lane);
  const v16bf aQ1 = load_a_frag_rm(Qp + (wid * 16) * HD + 32, HD, lane);

  v8f acc[4];  // 16 x 64 output accumulator
#pragma unroll
  for (int c = 0; c < 4; ++c) zero8(acc[c]);
  float mrow[8], lrow[8];
#pragma unroll
  for (int r = 0; r < 8; ++r) { mrow[r] = -INFINITY; lrow[r] = 0.0f; }

  const int n  = lane & 15;
  const int h2 = lane >> 4;
  const int nkb = qt * 2 + 2;   // causal: key blocks of 32 up to q0+63

  for (int jb = 0; jb < nkb; ++jb) {
    const int j0 = jb * 32;
    __syncthreads();
    // K tile row-major (async path): 32x64 = 256 chunks, 2 per thread.
#pragma unroll
    for (int v = tid; v < 256; v += 128) {
      const int row = v >> 3, part = v & 7;
      cp16(&Ks[row * HD + part * 8],
           &kb[base + (size_t)(j0 + row) * CDIM + part * 8]);
    }
    // V tile transposed: read 8 dims of one key, scatter to [dim][key].
#pragma unroll
    for (int v = tid; v < 256; v += 128) {
      const int key = v >> 3, part = v & 7;
      const uint4 d = *(const uint4*)&vb[base + (size_t)(j0 + key) * CDIM + part * 8];
      const u16* e = (const u16*)&d;
#pragma unroll
      for (int t = 0; t < 8; ++t)
        VsT[(part * 8 + t) * 32 + key] = e[t];
    }
    cp_wait();
    __syncthreads();

    const __bf16* Kp = reinterpret_cast<const __bf16*>(Ks);
    const __bf16* Vp = reinterpret_cast<const __bf16*>(VsT);

    // Scores: S[16q x 32k] as two 16x16 WMMA column tiles, K-dim 64 = 2 steps.
    // K^T fragment: element (k=16h+e, key n) = Ks[(s*16+n)*HD + kc*32 + 16h + e]
    // -> contiguous per lane.
    float Sv[2][8];
#pragma unroll
    for (int s = 0; s < 2; ++s) {
      v8f S;
      zero8(S);
      const v16bf bk0 = load_frag16(Kp + (s * 16 + n) * HD + 0 + 16 * h2);
      const v16bf bk1 = load_frag16(Kp + (s * 16 + n) * HD + 32 + 16 * h2);
      S = __builtin_amdgcn_wmma_f32_16x16x32_bf16(false, aQ0, false, bk0,
                                                  (short)0, S, false, false);
      S = __builtin_amdgcn_wmma_f32_16x16x32_bf16(false, aQ1, false, bk1,
                                                  (short)0, S, false, false);
#pragma unroll
      for (int r = 0; r < 8; ++r) {
        float x = S[r] * (float)HD;                 // reference quirk: *HEAD_DIM
        const int kg = j0 + s * 16 + n;
        const int qg = q0 + wid * 16 + r + 8 * h2;
        if (kg > qg) x = -INFINITY;                 // causal mask
        Sv[s][r] = x;
      }
    }

    // Online softmax update (row reductions across 16-lane halves).
#pragma unroll
    for (int r = 0; r < 8; ++r) {
      float bm = fmaxf(Sv[0][r], Sv[1][r]);
#pragma unroll
      for (int o = 8; o >= 1; o >>= 1) bm = fmaxf(bm, __shfl_xor(bm, o, 32));
      const float mnew = fmaxf(mrow[r], bm);
      const float sc = __expf(mrow[r] - mnew);      // first block: exp(-inf)=0
      mrow[r] = mnew;
      lrow[r] *= sc;
#pragma unroll
      for (int c = 0; c < 4; ++c) acc[c][r] *= sc;
      const float p0 = __expf(Sv[0][r] - mnew);
      const float p1 = __expf(Sv[1][r] - mnew);
      float rs = p0 + p1;
#pragma unroll
      for (int o = 8; o >= 1; o >>= 1) rs += __shfl_xor(rs, o, 32);
      lrow[r] += rs;
      const int m = r + 8 * h2;
      Ps[wid][m * 32 + n]      = f2bf(p0);
      Ps[wid][m * 32 + 16 + n] = f2bf(p1);
    }

    // O += P(16x32) @ V(32x64): per-wave P tile, no cross-wave barrier needed.
    const v16bf aP = load_a_frag_rm(reinterpret_cast<const __bf16*>(Ps[wid]), 32, lane);
#pragma unroll
    for (int c = 0; c < 4; ++c) {
      const v16bf bv = load_frag16(Vp + (c * 16 + n) * 32 + 16 * h2);
      acc[c] = __builtin_amdgcn_wmma_f32_16x16x32_bf16(false, aP, false, bv,
                                                       (short)0, acc[c], false, false);
    }
  }

  // Normalize and write bf16 (consumed by the O-projection GEMM).
#pragma unroll
  for (int r = 0; r < 8; ++r) {
    const float inv = 1.0f / lrow[r];
    const int qg = q0 + wid * 16 + r + 8 * h2;
#pragma unroll
    for (int c = 0; c < 4; ++c)
      ob[base + (size_t)qg * CDIM + c * 16 + n] = f2bf(acc[c][r] * inv);
  }
}

// ------------------------------------------------------------ launch -------

extern "C" void kernel_launch(void* const* d_in, const int* in_sizes, int n_in,
                              void* d_out, int out_size, void* d_ws, size_t ws_size,
                              hipStream_t stream) {
  const float* x  = (const float*)d_in[0];
  const float* Wk = (const float*)d_in[1];
  const float* Wq = (const float*)d_in[2];
  const float* Wv = (const float*)d_in[3];
  const float* Wo = (const float*)d_in[4];
  const float* W1 = (const float*)d_in[5];
  const float* W2 = (const float*)d_in[6];
  const float* g1 = (const float*)d_in[7];
  const float* b1 = (const float*)d_in[8];
  const float* g2 = (const float*)d_in[9];
  const float* b2 = (const float*)d_in[10];
  float* out = (float*)d_out;

  char* ws = (char*)d_ws;
  size_t off = 0;
  auto alloc = [&](size_t bytes) -> void* {
    void* p = ws + off;
    off += (bytes + 255) & ~(size_t)255;
    return p;
  };

  float* xn_f = (float*)alloc((size_t)BT * CDIM * 4);
  u16* xn_b   = (u16*)alloc((size_t)BT * CDIM * 2);
  u16* wq_b   = (u16*)alloc((size_t)CDIM * CDIM * 2);
  u16* wk_b   = (u16*)alloc((size_t)CDIM * CDIM * 2);
  u16* wv_b   = (u16*)alloc((size_t)CDIM * CDIM * 2);
  u16* wo_b   = (u16*)alloc((size_t)CDIM * CDIM * 2);
  u16* w1_b   = (u16*)alloc((size_t)CDIM * C4 * 2);
  u16* w2_b   = (u16*)alloc((size_t)C4 * CDIM * 2);
  u16* q_b    = (u16*)alloc((size_t)BT * CDIM * 2);
  u16* k_b    = (u16*)alloc((size_t)BT * CDIM * 2);
  u16* v_b    = (u16*)alloc((size_t)BT * CDIM * 2);
  u16* at_b   = (u16*)alloc((size_t)BT * CDIM * 2);
  float* h_f  = (float*)alloc((size_t)BT * CDIM * 4);
  float* h2_f = (float*)alloc((size_t)BT * CDIM * 4);
  u16* h2_b   = (u16*)alloc((size_t)BT * CDIM * 2);
  u16* a1_b   = (u16*)alloc((size_t)BT * C4 * 2);

  auto conv = [&](const float* src, u16* dst, size_t nElem) {
    const int n4 = (int)(nElem / 4);
    f2bf4_kernel<<<dim3((unsigned)((n4 + 255) / 256)), dim3(256), 0, stream>>>(
        (const float4*)src, (ushort4*)dst, n4);
  };
  conv(Wq, wq_b, (size_t)CDIM * CDIM);
  conv(Wk, wk_b, (size_t)CDIM * CDIM);
  conv(Wv, wv_b, (size_t)CDIM * CDIM);
  conv(Wo, wo_b, (size_t)CDIM * CDIM);
  conv(W1, w1_b, (size_t)CDIM * C4);
  conv(W2, w2_b, (size_t)C4 * CDIM);

  // xn = LN1(x)
  ln_kernel<<<dim3(BT), dim3(256), 0, stream>>>(x, g1, b1, xn_f, xn_b);

  // Q/K/V projections: write bf16 activations only.
  const dim3 gc(BT / 128, CDIM / 128);
  gemm_bf16_kernel<false, true, false, false><<<gc, 256, 0, stream>>>(
      xn_b, wq_b, nullptr, nullptr, q_b, BT, CDIM, CDIM);
  gemm_bf16_kernel<false, true, false, false><<<gc, 256, 0, stream>>>(
      xn_b, wk_b, nullptr, nullptr, k_b, BT, CDIM, CDIM);
  gemm_bf16_kernel<false, true, false, false><<<gc, 256, 0, stream>>>(
      xn_b, wv_b, nullptr, nullptr, v_b, BT, CDIM, CDIM);

  // causal attention
  attn_kernel<<<dim3(TLEN / 64, NHEAD, BSZ), 128, 0, stream>>>(q_b, k_b, v_b, at_b);

  // h = xn + attn @ Wo  (fp32 out)
  gemm_bf16_kernel<true, false, true, false><<<gc, 256, 0, stream>>>(
      at_b, wo_b, xn_f, h_f, nullptr, BT, CDIM, CDIM);

  // h2 = LN2(h)
  ln_kernel<<<dim3(BT), dim3(256), 0, stream>>>(h_f, g2, b2, h2_f, h2_b);

  // a1 = relu(h2 @ W1)  (bf16 out)
  const dim3 gm1(BT / 128, C4 / 128);
  gemm_bf16_kernel<false, true, false, true><<<gm1, 256, 0, stream>>>(
      h2_b, w1_b, nullptr, nullptr, a1_b, BT, C4, CDIM);

  // out = h2 + a1 @ W2   (source quirk: residual uses LN2 output)
  gemm_bf16_kernel<true, false, true, false><<<gc, 256, 0, stream>>>(
      a1_b, w2_b, h2_f, out, nullptr, BT, CDIM, C4);
}
